// LSTMClassifier_1391569404593
// MI455X (gfx1250) — matrix operand order repaired and checked by reading, not compiled, not run on hardware
//
#include <hip/hip_runtime.h>
#include <hip/hip_bf16.h>
#include <math.h>

// ---------------------------------------------------------------------------
// LSTM classifier for MI455X (gfx1250), wave32 + WMMA bf16, single persistent
// kernel for the whole T=256 recurrence.
//   V=50000, E=512, H=1024, O=2, B=64, T=256, PAD=1
// Per step:  z[64,4096] = [xt|h][64,1536] @ [W_all;U_all][1536,4096]  (WMMA)
//            -> grid barrier -> fused gates (c in regs, v_exp/v_rcp
//               transcendentals) -> grid barrier.
// 32 blocks x 256 threads = 256 waves = exactly the 4x64 strip grid; weights
// bf16 and L2-resident (12 MB << 192 MB L2).
// ---------------------------------------------------------------------------

#define BB   64
#define TT   256
#define EE   512
#define HH   1024
#define H4   4096
#define PAD_IDX 1
#define NBLK 32

typedef __attribute__((ext_vector_type(16))) __bf16 v16bf;
typedef __attribute__((ext_vector_type(8)))  __bf16 v8bf;
typedef __attribute__((ext_vector_type(8)))  float  v8f;

__device__ __forceinline__ v16bf cat8(v8bf a, v8bf b) {
    return __builtin_shufflevector(a, b,
        0, 1, 2, 3, 4, 5, 6, 7, 8, 9, 10, 11, 12, 13, 14, 15);
}

// Hardware-transcendental activations: v_exp_f32 + v_rcp_f32 (TRANS pipe).
__device__ __forceinline__ float fast_sigmoid(float x) {
    return __builtin_amdgcn_rcpf(1.f + __expf(-x));
}
__device__ __forceinline__ float fast_tanh(float x) {
    // tanh(x) = 1 - 2/(exp(2x)+1); saturates correctly at +/-inf.
    return 1.f - 2.f * __builtin_amdgcn_rcpf(__expf(2.f * x) + 1.f);
}

// Branch-free K-phase: A rows from aRow (contiguous K), B from bMat[K][4H].
// Accumulates 4 adjacent 16x16 N-tiles sharing one A fragment.
__device__ __forceinline__ void gemm_phase(const __bf16* __restrict__ aRow,
                                           const __bf16* __restrict__ bMat,
                                           int kCount, int n0, int lane,
                                           v8f acc[4]) {
    const int hh = lane >> 4;                   // lane half (ISA A layout)
    #pragma unroll 2
    for (int kk = 0; kk < kCount; kk += 32) {
        // A fragment (16x32 bf16): lanes0-15 K in [0,8)+[16,24);
        // lanes16-31 K in [8,16)+[24,32).
        v8bf a0 = *(const v8bf*)(aRow + kk + 8 * hh);
        v8bf a1 = *(const v8bf*)(aRow + kk + 16 + 8 * hh);
        v16bf af = cat8(a0, a1);

        // B fragments: lane holds column n = n0 + 16j + lane%16; K in same order as A.
        const __bf16* bRow = bMat + (size_t)(kk + 8 * hh) * H4 + n0 + (lane & 15);
        #pragma unroll
        for (int j = 0; j < 4; ++j) {
            v16bf bf;
            #pragma unroll
            for (int s = 0; s < 8; ++s) {
                bf[s]     = bRow[(size_t)s * H4 + 16 * j];
                bf[8 + s] = bRow[(size_t)(16 + s) * H4 + 16 * j];
            }
            acc[j] = __builtin_amdgcn_wmma_f32_16x16x32_bf16(
                false, af, false, bf,
                (short)0, acc[j], false, false);
        }
    }
}

// Grid-wide barrier: monotonically increasing arrive counter, device-scope
// atomics, s_sleep spin. Safe: 32 co-resident blocks.
__device__ __forceinline__ void grid_barrier(int* bar, int idx) {
    __threadfence();                 // my stores -> device scope
    __syncthreads();                 // whole block arrived
    if (threadIdx.x == 0) {
        __hip_atomic_fetch_add(bar, 1, __ATOMIC_ACQ_REL,
                               __HIP_MEMORY_SCOPE_AGENT);
        while (__hip_atomic_load(bar, __ATOMIC_ACQUIRE,
                                 __HIP_MEMORY_SCOPE_AGENT) < idx * NBLK)
            __builtin_amdgcn_s_sleep(1);
    }
    __syncthreads();
    __threadfence();                 // discard stale near-cache lines
}

// ---------------- fp32 -> bf16 weight conversion ---------------------------
__global__ void f32_to_bf16_kernel(const float* __restrict__ s,
                                   __bf16* __restrict__ d, int n) {
    int i = blockIdx.x * 256 + threadIdx.x;
    if (i < n) d[i] = (__bf16)s[i];
}

// ---------------- embedding gather (fp32 table -> bf16 activations) --------
__global__ void embed_kernel(const int* __restrict__ x,
                             const float* __restrict__ W_emb,
                             __bf16* __restrict__ embB) {
    long i = (long)blockIdx.x * 256 + threadIdx.x;      // over B*T*E
    if (i >= (long)BB * TT * EE) return;
    long bt = i / EE;
    int  e  = (int)(i % EE);
    int  tok = x[bt];
    embB[i] = (__bf16)W_emb[(long)tok * EE + e];
}

// ---------------- zero h / finalH / barrier counter ------------------------
__global__ void init_state_kernel(__bf16* __restrict__ hB,
                                  float* __restrict__ finalH,
                                  int* __restrict__ bar) {
    int i = blockIdx.x * 256 + threadIdx.x;
    if (i < BB * HH) { hB[i] = (__bf16)0.f; finalH[i] = 0.f; }
    if (i == 0) *bar = 0;
}

// ---------------- per-batch length / pick index / pad flag -----------------
__global__ void meta_kernel(const int* __restrict__ x,
                            int* __restrict__ tpick,
                            int* __restrict__ padflag) {
    int b = threadIdx.x;
    if (b >= BB) return;
    int len = 0;
    for (int t = 0; t < TT; ++t) len += (x[b * TT + t] != PAD_IDX);
    int tp = (len > 0) ? (len - 1) : 0;
    tpick[b]   = tp;
    padflag[b] = (x[b * TT + tp] == PAD_IDX) ? 1 : 0;
}

// ---------------- persistent recurrence: all 256 LSTM steps ----------------
__global__ __launch_bounds__(256) void lstm_persistent(
        const __bf16* __restrict__ embB,   // [B, T, E]
        __bf16* __restrict__ hB,           // [B, H]   (read+write)
        const __bf16* __restrict__ Wb,     // [E, 4H]
        const __bf16* __restrict__ Ub,     // [H, 4H]
        float* __restrict__ z,             // [B, 4H]
        const float* __restrict__ b_all,   // [4H]
        float* __restrict__ finalH,        // [B, H]
        const int* __restrict__ tpick, const int* __restrict__ padflag,
        int* __restrict__ bar) {
    // ---- GEMM-phase geometry: one wave per 16x64 output strip -------------
    const int tid  = threadIdx.x;
    const int lane = tid & 31;
    const int wave = tid >> 5;
    const int strip = blockIdx.x * 8 + wave;    // 0..255
    const int m0 = (strip & 3) << 4;            // 0,16,32,48
    const int n0 = (strip >> 2) << 6;           // 0..4032 step 64
    const int l  = lane & 15;
    const int hh = lane >> 4;
    const int rowA = m0 + l;                    // batch row for A loads
    const __bf16* embRow0 = embB + (size_t)rowA * TT * EE;
    const __bf16* hRow    = hB + (size_t)rowA * HH;

    // ---- gate-phase geometry: 8 contiguous (b,k) cells per thread ---------
    const int gtid = blockIdx.x * 256 + tid;    // 0..8191
    const int j0 = gtid * 8;                    // over B*H = 65536
    const int gb = j0 >> 10;
    const int gk = j0 & (HH - 1);
    float creg[8];                              // fp32 cell state, in regs
    float bia[8], bfa[8], bga[8], boa[8];       // biases, preloaded
    #pragma unroll
    for (int r = 0; r < 8; ++r) {
        creg[r] = 0.f;
        bia[r] = b_all[gk + r];
        bfa[r] = b_all[HH + gk + r];
        bga[r] = b_all[2 * HH + gk + r];
        boa[r] = b_all[3 * HH + gk + r];
    }
    const int tp = tpick[gb];
    const int pf = padflag[gb];
    const float* zrow   = z + (size_t)gb * H4 + gk;
    __bf16*      hOut   = hB + j0;
    float*       fHOut  = finalH + j0;

    int barIdx = 0;
    for (int t = 0; t < TT; ++t) {
        // ---- phase 1: z = [x_t | h] @ [Wb; Ub] ----------------------------
        v8f acc[4] = {{}, {}, {}, {}};
        gemm_phase(embRow0 + (size_t)t * EE, Wb, EE, n0, lane, acc);
        gemm_phase(hRow,                     Ub, HH, n0, lane, acc);
        #pragma unroll
        for (int j = 0; j < 4; ++j)
            #pragma unroll
            for (int r = 0; r < 8; ++r)
                z[(size_t)(m0 + r + 8 * hh) * H4 + n0 + 16 * j + l] = acc[j][r];

        grid_barrier(bar, ++barIdx);            // z visible everywhere

        // ---- phase 2: fused gates, c in registers -------------------------
        #pragma unroll
        for (int r = 0; r < 8; ++r) {
            float zi = zrow[r]           + bia[r];
            float zf = zrow[HH + r]      + bfa[r];
            float zg = zrow[2 * HH + r]  + bga[r];
            float zo = zrow[3 * HH + r]  + boa[r];
            float cn = fast_sigmoid(zf) * creg[r]
                     + fast_sigmoid(zi) * fast_tanh(zg);
            float hn = fast_sigmoid(zo) * fast_tanh(cn);
            creg[r] = cn;
            hOut[r] = (__bf16)hn;
            if (t == tp) fHOut[r] = pf ? 0.f : hn;
        }

        grid_barrier(bar, ++barIdx);            // h visible before next GEMM
    }
}

// ---------------- tiny output projection [64,1024] @ [1024,2] --------------
__global__ void proj_kernel(const float* __restrict__ finalH,
                            const float* __restrict__ W_out,
                            const float* __restrict__ b_out,
                            float* __restrict__ out) {
    int idx = threadIdx.x;                      // 128 = B*O
    if (idx >= BB * 2) return;
    int b = idx >> 1, o = idx & 1;
    float s = b_out[o];
    for (int k = 0; k < HH; ++k)
        s += finalH[(size_t)b * HH + k] * W_out[k * 2 + o];
    out[idx] = s;
}

// ---------------------------------------------------------------------------
extern "C" void kernel_launch(void* const* d_in, const int* in_sizes, int n_in,
                              void* d_out, int out_size, void* d_ws, size_t ws_size,
                              hipStream_t stream) {
    (void)in_sizes; (void)n_in; (void)out_size; (void)ws_size;
    const int*   x     = (const int*)d_in[0];
    const float* W_emb = (const float*)d_in[1];
    const float* W_all = (const float*)d_in[2];
    const float* U_all = (const float*)d_in[3];
    const float* b_all = (const float*)d_in[4];
    const float* W_out = (const float*)d_in[5];
    const float* b_out = (const float*)d_in[6];

    // workspace layout (~30 MB)
    char* w = (char*)d_ws;
    __bf16* Wb    = (__bf16*)(w);                                //  4 MiB
    __bf16* Ub    = (__bf16*)(w + (4ull  << 20));                //  8 MiB
    __bf16* embB  = (__bf16*)(w + (12ull << 20));                // 16 MiB
    float*  z     = (float*) (w + (28ull << 20));                //  1 MiB
    __bf16* hB    = (__bf16*)(w + (29ull << 20));                // 128 KiB
    float*  fH    = (float*) (w + (29ull << 20) + (128u << 10)); // 256 KiB
    int*    tpick = (int*)   (w + (29ull << 20) + (384u << 10));
    int*    padfl = tpick + BB;
    int*    bar   = padfl + BB;

    // prologue: weight conversion, embedding gather, state init, pad meta
    f32_to_bf16_kernel<<<(EE * H4 + 255) / 256, 256, 0, stream>>>(W_all, Wb, EE * H4);
    f32_to_bf16_kernel<<<(HH * H4 + 255) / 256, 256, 0, stream>>>(U_all, Ub, HH * H4);
    embed_kernel<<<(BB * TT * EE + 255) / 256, 256, 0, stream>>>(x, W_emb, embB);
    init_state_kernel<<<(BB * HH + 255) / 256, 256, 0, stream>>>(hB, fH, bar);
    meta_kernel<<<1, 64, 0, stream>>>(x, tpick, padfl);

    // entire recurrence: ONE persistent kernel (grid barriers between phases)
    lstm_persistent<<<NBLK, 256, 0, stream>>>(embB, hB, Wb, Ub, z, b_all,
                                              fH, tpick, padfl, bar);

    // epilogue: projection to [64, 2]
    proj_kernel<<<1, 128, 0, stream>>>(fH, W_out, b_out, (float*)d_out);
}
